// SlidingWindowAttention_18657337934234
// MI455X (gfx1250) — matrix operand order repaired and checked by reading, not compiled, not run on hardware
//
#include <hip/hip_runtime.h>
#include <hip/hip_bf16.h>
#include <stdint.h>

// ---- problem constants (from reference) ----
#define BB   2
#define SS   4096
#define DD   2048
#define HQn  16
#define HKVn 4
#define HDn  128
#define WW   512
#define NROWS (BB*SS)          // 8192
#define EPSf 1e-6f

typedef __attribute__((ext_vector_type(16))) __bf16 v16bf;
typedef __attribute__((ext_vector_type(8)))  float  v8f;

union FragA { v16bf v; uint32_t u[8]; uint4 q[2]; };
union FragC { v8f  v; float    f[8]; };

__device__ __forceinline__ uint16_t f2bf(float x){
    uint32_t u = __float_as_uint(x);
    uint32_t r = u + 0x7FFFu + ((u >> 16) & 1u);   // round-to-nearest-even
    return (uint16_t)(r >> 16);
}
__device__ __forceinline__ float bf2f(uint16_t x){
    return __uint_as_float(((uint32_t)x) << 16);
}

// CDNA5 async global->LDS copy of 16 bytes (ASYNCcnt-tracked, no VGPR data path).
__device__ __forceinline__ void async_b128(const void* gp, void* lp){
    uint32_t lds = (uint32_t)(uintptr_t)lp;           // low 32 bits of generic = LDS offset
    uint64_t ga  = (uint64_t)(uintptr_t)gp;
    asm volatile("global_load_async_to_lds_b128 %0, %1, off"
                 :: "v"(lds), "v"(ga) : "memory");
}
__device__ __forceinline__ void async_wait0(){
    asm volatile("s_wait_asynccnt 0x0" ::: "memory");
}

// pack two row-vectors of 8 bf16 into 8 k-pair dwords, store transposed to LDS
__device__ __forceinline__ void pack_store_tr(uint4 r0, uint4 r1,
                                              uint16_t* __restrict__ base,
                                              int tc, int tk){
    const uint16_t* e0 = (const uint16_t*)&r0;
    const uint16_t* e1 = (const uint16_t*)&r1;
    #pragma unroll
    for (int j = 0; j < 8; ++j){
        uint32_t dw = (uint32_t)e0[j] | ((uint32_t)e1[j] << 16);
        *(uint32_t*)&base[(tc + j) * 32 + tk] = dw;
    }
}

// ============================ RMSNorm -> bf16 ============================
__global__ __launch_bounds__(256)
void rmsnorm_kernel(const float* __restrict__ x, const float* __restrict__ g,
                    uint16_t* __restrict__ xn){
    int row = blockIdx.x;
    const float* xr = x + (size_t)row * DD;
    __shared__ float red[256];
    float ss = 0.f;
    for (int c = threadIdx.x; c < DD; c += 256){ float v = xr[c]; ss += v*v; }
    red[threadIdx.x] = ss; __syncthreads();
    for (int off = 128; off > 0; off >>= 1){
        if (threadIdx.x < off) red[threadIdx.x] += red[threadIdx.x + off];
        __syncthreads();
    }
    float rms = rsqrtf(red[0] / (float)DD + EPSf);
    uint16_t* o = xn + (size_t)row * DD;
    for (int c = threadIdx.x; c < DD; c += 256) o[c] = f2bf(xr[c] * rms * g[c]);
}

// ============================ f32 -> bf16 copy ============================
__global__ __launch_bounds__(256)
void cvt_bf16_kernel(const float* __restrict__ in, uint16_t* __restrict__ out, int n){
    int i = blockIdx.x * 256 + threadIdx.x;
    if (i < n) out[i] = f2bf(in[i]);
}

// ============================ bf16 WMMA GEMM (double-buffered) ============================
// C[M,N] = A[M,K] @ B[K,N]; A,B bf16 row-major; C f32 or bf16 (out_bf16).
// Block tile 64x128, 8 waves = (rowtile 0..3) x (col half 0..1), 16x64 per wave.
// Ping-pong LDS: next A tile via async global->LDS, next B rows via global b128,
// packed/stored transposed AFTER the current tile's WMMAs; one barrier per K-step.
__global__ __launch_bounds__(256)
void gemm_bf16_kernel(const uint16_t* __restrict__ A, const uint16_t* __restrict__ Bm,
                      void* __restrict__ Cout, int M, int N, int K, int out_bf16){
    __shared__ __align__(16) uint16_t As[2][64*32];     // row-major [m][k]
    __shared__ __align__(16) uint16_t Bt[2][128*32];    // transposed [n][k]
    const int mbase = blockIdx.x * 64, nbase = blockIdx.y * 128;
    const int tid = threadIdx.x, wave = tid >> 5, lane = tid & 31;
    const int rt = wave & 3, ch = wave >> 2;

    FragC acc[4];
    #pragma unroll
    for (int t = 0; t < 4; ++t)
        #pragma unroll
        for (int r = 0; r < 8; ++r) acc[t].f[r] = 0.f;

    const int arow = tid >> 2, acol = (tid & 3) * 8;      // A: 64x32, one b128/thread
    const int tk = (tid & 15) * 2, tc = (tid >> 4) * 8;   // B: k-pair x col-group

    // ---- prologue: stage tile k0=0 into buffer 0 ----
    async_b128(&A[(size_t)(mbase + arow) * K + acol], &As[0][arow*32 + acol]);
    {
        uint4 r0 = *(const uint4*)&Bm[(size_t)(tk)     * N + nbase + tc];
        uint4 r1 = *(const uint4*)&Bm[(size_t)(tk + 1) * N + nbase + tc];
        pack_store_tr(r0, r1, Bt[0], tc, tk);
    }
    async_wait0();
    __syncthreads();

    int bsel = 0;
    for (int k0 = 0; k0 < K; k0 += 32){
        const int nxt = bsel ^ 1;
        const bool more = (k0 + 32 < K);
        uint4 nb0 = {}, nb1 = {};
        if (more){
            async_b128(&A[(size_t)(mbase + arow) * K + k0 + 32 + acol],
                       &As[nxt][arow*32 + acol]);
            nb0 = *(const uint4*)&Bm[(size_t)(k0 + 32 + tk)     * N + nbase + tc];
            nb1 = *(const uint4*)&Bm[(size_t)(k0 + 32 + tk + 1) * N + nbase + tc];
            if (k0 + 64 < K)
                __builtin_prefetch(&Bm[(size_t)(k0 + 64 + tk) * N + nbase + tc], 0, 0);
        }

        // ---- compute current tile: preload all fragments, then 4 WMMAs ----
        const uint16_t* Asb = As[bsel];
        const uint16_t* Btb = Bt[bsel];
        FragA af;
        {
            const int r = lane & 15, kb = (lane < 16) ? 0 : 8;
            af.q[0] = *(const uint4*)&Asb[(rt*16 + r) * 32 + kb];
            af.q[1] = *(const uint4*)&Asb[(rt*16 + r) * 32 + kb + 16];
        }
        FragA bfr[4];
        {
            const int kbb = (lane < 16) ? 0 : 8;
            #pragma unroll
            for (int t = 0; t < 4; ++t){
                const int n = ch*64 + t*16 + (lane & 15);
                bfr[t].q[0] = *(const uint4*)&Btb[n*32 + kbb];
                bfr[t].q[1] = *(const uint4*)&Btb[n*32 + kbb + 16];
            }
        }
        #pragma unroll
        for (int t = 0; t < 4; ++t)
            acc[t].v = __builtin_amdgcn_wmma_f32_16x16x32_bf16(
                false, af.v, false, bfr[t].v, (short)0, acc[t].v, false, false);

        // ---- finish staging next tile ----
        if (more){
            pack_store_tr(nb0, nb1, Bt[nxt], tc, tk);
            async_wait0();
        }
        __syncthreads();
        bsel = nxt;
    }

    const int rrow = mbase + rt*16 + ((lane >> 4) & 1) * 8;
    #pragma unroll
    for (int t = 0; t < 4; ++t){
        const int col = nbase + ch*64 + t*16 + (lane & 15);
        #pragma unroll
        for (int r = 0; r < 8; ++r){
            if (out_bf16) ((uint16_t*)Cout)[(size_t)(rrow + r) * N + col] = f2bf(acc[t].f[r]);
            else          ((float*)Cout)[(size_t)(rrow + r) * N + col]    = acc[t].f[r];
        }
    }
}

// ============================ RoPE + head-major relayout ============================
__global__ __launch_bounds__(256)
void rope_cvt_kernel(const uint16_t* __restrict__ in, uint16_t* __restrict__ out,
                     int nh, int do_rope){
    const int row = blockIdx.x;
    const int b = row / SS, s = row % SS;
    const uint16_t* ir = in + (size_t)row * nh * HDn;
    const int npairs = nh * HDn / 2;
    for (int p = threadIdx.x; p < npairs; p += 256){
        const int h = p / (HDn/2), i = p % (HDn/2);
        float xe = bf2f(ir[h*HDn + 2*i]), xo = bf2f(ir[h*HDn + 2*i + 1]);
        float oe = xe, oo = xo;
        if (do_rope){
            float inv = __powf(10000.f, -2.f * (float)i / (float)HDn);
            float ang = (float)s * inv;
            float sn, cs; __sincosf(ang, &sn, &cs);
            oe = xe*cs - xo*sn; oo = xe*sn + xo*cs;
        }
        size_t ob = ((size_t)(b*nh + h) * SS + s) * HDn;
        out[ob + 2*i]     = f2bf(oe);
        out[ob + 2*i + 1] = f2bf(oo);
    }
}

// ============================ Flash sliding-window attention (double-buffered) ============================
// grid.x = B*HQ*(S/128); workgroup: 128 queries of one (b,h); wave w owns 16 queries.
// Ping-pong LDS: next K tile via async DMA (row-major feeds Q@K^T B-fragments
// directly), next V rows via global b128 packed/stored transposed after compute.
__global__ __launch_bounds__(256)
void swa_kernel(const uint16_t* __restrict__ Q, const uint16_t* __restrict__ Km,
                const uint16_t* __restrict__ Vm, uint16_t* __restrict__ Out){
    __shared__ __align__(16) uint16_t Ks[2][32*128];    // row-major [key][feat]
    __shared__ __align__(16) uint16_t Vt[2][128*32];    // transposed [feat][key]
    __shared__ __align__(16) uint16_t Ps[8][16*32];

    const int wg = blockIdx.x;
    const int mslice = wg % (SS/128);
    const int bh = wg / (SS/128);
    const int h = bh % HQn, b = bh / HQn;
    const int kvh = h / (HQn / HKVn);
    const int qbase = mslice * 128;
    const int tid = threadIdx.x, wave = tid >> 5, lane = tid & 31;
    const int qw = qbase + wave * 16;
    const int myrow_off = (lane >> 4) * 8;

    const uint16_t* Qb = Q  + ((size_t)(b*HQn  + h  ) * SS) * HDn;
    const uint16_t* Kb = Km + ((size_t)(b*HKVn + kvh) * SS) * HDn;
    const uint16_t* Vb = Vm + ((size_t)(b*HKVn + kvh) * SS) * HDn;

    // Q fragments: 4 K-chunks of 32 features, 2x b128 global loads each
    FragA qf[4];
    {
        const int r = lane & 15, kb = (lane < 16) ? 0 : 8;
        #pragma unroll
        for (int c = 0; c < 4; ++c){
            const size_t base = (size_t)(qw + r) * HDn + c*32 + kb;
            qf[c].q[0] = *(const uint4*)&Qb[base];
            qf[c].q[1] = *(const uint4*)&Qb[base + 16];
        }
    }

    FragC o[8];
    #pragma unroll
    for (int t = 0; t < 8; ++t)
        #pragma unroll
        for (int r = 0; r < 8; ++r) o[t].f[r] = 0.f;
    float mrow[8], lrow[8], alpha[8];
    #pragma unroll
    for (int r = 0; r < 8; ++r){ mrow[r] = -1e30f; lrow[r] = 0.f; }

    int kstart = qbase - WW; if (kstart < 0) kstart = 0;
    const int kend = qbase + 128;
    const float scale = 0.08838834764831845f;  // 1/sqrt(128)

    const int krr = tid >> 3, kcc = (tid & 7) * 16;     // K staging coords
    const int tk = (tid & 15) * 2, tc = (tid >> 4) * 8; // V transpose coords

    // ---- prologue: stage tile kstart into buffer 0 ----
    async_b128(&Kb[(size_t)(kstart + krr) * HDn + kcc],     &Ks[0][krr*128 + kcc]);
    async_b128(&Kb[(size_t)(kstart + krr) * HDn + kcc + 8], &Ks[0][krr*128 + kcc + 8]);
    {
        uint4 r0 = *(const uint4*)&Vb[(size_t)(kstart + tk)     * HDn + tc];
        uint4 r1 = *(const uint4*)&Vb[(size_t)(kstart + tk + 1) * HDn + tc];
        pack_store_tr(r0, r1, Vt[0], tc, tk);
    }
    async_wait0();
    __syncthreads();

    int bsel = 0;
    for (int kt = kstart; kt < kend; kt += 32){
        const int nxt = bsel ^ 1;
        const bool more = (kt + 32 < kend);
        uint4 nv0 = {}, nv1 = {};
        if (more){
            async_b128(&Kb[(size_t)(kt + 32 + krr) * HDn + kcc],
                       &Ks[nxt][krr*128 + kcc]);
            async_b128(&Kb[(size_t)(kt + 32 + krr) * HDn + kcc + 8],
                       &Ks[nxt][krr*128 + kcc + 8]);
            nv0 = *(const uint4*)&Vb[(size_t)(kt + 32 + tk)     * HDn + tc];
            nv1 = *(const uint4*)&Vb[(size_t)(kt + 32 + tk + 1) * HDn + tc];
        }
        const uint16_t* Ksb = Ks[bsel];
        const uint16_t* Vtb = Vt[bsel];

        // ---- S = Q @ K^T -> two 16x16 tiles; fragments preloaded in groups of 4 ----
        FragC s0, s1;
        #pragma unroll
        for (int r = 0; r < 8; ++r){ s0.f[r] = 0.f; s1.f[r] = 0.f; }
        {
            const int col = lane & 15, kbb = (lane < 16) ? 0 : 8;
            #pragma unroll
            for (int g = 0; g < 2; ++g){          // c-chunk pairs {0,1}, {2,3}
                FragA bb[4];
                #pragma unroll
                for (int cc = 0; cc < 2; ++cc){
                    const int c = g*2 + cc;
                    bb[cc*2+0].q[0] = *(const uint4*)&Ksb[col * 128 + c*32 + kbb];
                    bb[cc*2+0].q[1] = *(const uint4*)&Ksb[col * 128 + c*32 + kbb + 16];
                    bb[cc*2+1].q[0] = *(const uint4*)&Ksb[(16 + col) * 128 + c*32 + kbb];
                    bb[cc*2+1].q[1] = *(const uint4*)&Ksb[(16 + col) * 128 + c*32 + kbb + 16];
                }
                s0.v = __builtin_amdgcn_wmma_f32_16x16x32_bf16(
                    false, qf[g*2+0].v, false, bb[0].v, (short)0, s0.v, false, false);
                s1.v = __builtin_amdgcn_wmma_f32_16x16x32_bf16(
                    false, qf[g*2+0].v, false, bb[1].v, (short)0, s1.v, false, false);
                s0.v = __builtin_amdgcn_wmma_f32_16x16x32_bf16(
                    false, qf[g*2+1].v, false, bb[2].v, (short)0, s0.v, false, false);
                s1.v = __builtin_amdgcn_wmma_f32_16x16x32_bf16(
                    false, qf[g*2+1].v, false, bb[3].v, (short)0, s1.v, false, false);
            }
        }

        // ---- mask + online softmax (row stats via 16-lane shfl reductions) ----
        const int kj0 = kt + (lane & 15);
        #pragma unroll
        for (int r = 0; r < 8; ++r){
            const int qi = qw + myrow_off + r;
            bool m0 = (kj0 > qi) || (kj0 < qi - WW);
            bool m1 = (kj0 + 16 > qi) || (kj0 + 16 < qi - WW);
            float v0 = m0 ? -1e30f : s0.f[r] * scale;
            float v1 = m1 ? -1e30f : s1.f[r] * scale;
            float x = fmaxf(v0, v1);
            #pragma unroll
            for (int off = 1; off < 16; off <<= 1) x = fmaxf(x, __shfl_xor(x, off, 16));
            float mn = fmaxf(mrow[r], x);
            float al = __expf(mrow[r] - mn);
            mrow[r] = mn;
            float p0 = m0 ? 0.f : __expf(v0 - mn);
            float p1 = m1 ? 0.f : __expf(v1 - mn);
            s0.f[r] = p0; s1.f[r] = p1;
            float sm = p0 + p1;
            #pragma unroll
            for (int off = 1; off < 16; off <<= 1) sm += __shfl_xor(sm, off, 16);
            lrow[r] = lrow[r] * al + sm;
            alpha[r] = al;
        }
        #pragma unroll
        for (int t = 0; t < 8; ++t)
            #pragma unroll
            for (int r = 0; r < 8; ++r) o[t].f[r] *= alpha[r];

        // ---- P (C-layout) -> per-wave LDS -> A-fragment (16x32) ----
        {
            uint16_t* pp = Ps[wave];
            const int colb = lane & 15;
            #pragma unroll
            for (int r = 0; r < 8; ++r){
                pp[(myrow_off + r) * 32 + colb]      = f2bf(s0.f[r]);
                pp[(myrow_off + r) * 32 + 16 + colb] = f2bf(s1.f[r]);
            }
        }
        FragA pf;
        {
            const uint16_t* pp = Ps[wave];
            const int r = lane & 15, kb = (lane < 16) ? 0 : 8;
            pf.q[0] = *(const uint4*)&pp[r * 32 + kb];
            pf.q[1] = *(const uint4*)&pp[r * 32 + kb + 16];
        }
        // ---- O += P @ V; fragments preloaded in groups of 4 ----
        {
            const int colf = lane & 15, kbb = (lane < 16) ? 0 : 8;
            #pragma unroll
            for (int g = 0; g < 2; ++g){
                FragA vf4[4];
                #pragma unroll
                for (int j = 0; j < 4; ++j){
                    const int f = (g*4 + j)*16 + colf;
                    vf4[j].q[0] = *(const uint4*)&Vtb[f*32 + kbb];
                    vf4[j].q[1] = *(const uint4*)&Vtb[f*32 + kbb + 16];
                }
                #pragma unroll
                for (int j = 0; j < 4; ++j)
                    o[g*4+j].v = __builtin_amdgcn_wmma_f32_16x16x32_bf16(
                        false, pf.v, false, vf4[j].v, (short)0, o[g*4+j].v, false, false);
            }
        }

        // ---- finish staging next tile ----
        if (more){
            pack_store_tr(nv0, nv1, Vt[nxt], tc, tk);
            async_wait0();
        }
        __syncthreads();
        bsel = nxt;
    }

    // normalize + store bf16 to attn buffer [B*S, D] (col = h*HD + d)
    #pragma unroll
    for (int t = 0; t < 8; ++t)
        #pragma unroll
        for (int r = 0; r < 8; ++r){
            const int qi = qw + myrow_off + r;
            float val = o[t].f[r] / lrow[r];
            Out[(size_t)(b*SS + qi) * DD + h*HDn + t*16 + (lane & 15)] = f2bf(val);
        }
}

// ============================ host-side launch ============================
extern "C" void kernel_launch(void* const* d_in, const int* in_sizes, int n_in,
                              void* d_out, int out_size, void* d_ws, size_t ws_size,
                              hipStream_t stream){
    (void)in_sizes; (void)n_in; (void)out_size; (void)ws_size;
    const float* x  = (const float*)d_in[0];
    const float* g  = (const float*)d_in[1];
    const float* wq = (const float*)d_in[2];
    const float* wk = (const float*)d_in[3];
    const float* wv = (const float*)d_in[4];
    const float* wo = (const float*)d_in[5];

    char* ws = (char*)d_ws;
    size_t off = 0;
    auto alloc = [&](size_t bytes) -> char* {
        char* p = ws + off;
        off += (bytes + 255) & ~(size_t)255;
        return p;
    };
    uint16_t* xn   = (uint16_t*)alloc((size_t)NROWS * DD * 2);        // 32 MB
    uint16_t* wqb  = (uint16_t*)alloc((size_t)DD * DD * 2);           //  8 MB
    uint16_t* wkb  = (uint16_t*)alloc((size_t)DD * 512 * 2);          //  2 MB
    uint16_t* wvb  = (uint16_t*)alloc((size_t)DD * 512 * 2);          //  2 MB
    uint16_t* wob  = (uint16_t*)alloc((size_t)DD * DD * 2);           //  8 MB
    uint16_t* qraw = (uint16_t*)alloc((size_t)NROWS * 2048 * 2);      // 32 MB
    uint16_t* kraw = (uint16_t*)alloc((size_t)NROWS * 512 * 2);       //  8 MB
    uint16_t* vraw = (uint16_t*)alloc((size_t)NROWS * 512 * 2);       //  8 MB
    uint16_t* qhm  = (uint16_t*)alloc((size_t)BB*HQn *SS*HDn * 2);    // 32 MB
    uint16_t* khm  = (uint16_t*)alloc((size_t)BB*HKVn*SS*HDn * 2);    //  8 MB
    uint16_t* vhm  = (uint16_t*)alloc((size_t)BB*HKVn*SS*HDn * 2);    //  8 MB
    uint16_t* attn = (uint16_t*)alloc((size_t)NROWS * DD * 2);        // 32 MB

    // 1. RMSNorm -> bf16
    rmsnorm_kernel<<<NROWS, 256, 0, stream>>>(x, g, xn);

    // 2. weights -> bf16
    cvt_bf16_kernel<<<(DD*DD + 255)/256, 256, 0, stream>>>(wq, wqb, DD*DD);
    cvt_bf16_kernel<<<(DD*512 + 255)/256, 256, 0, stream>>>(wk, wkb, DD*512);
    cvt_bf16_kernel<<<(DD*512 + 255)/256, 256, 0, stream>>>(wv, wvb, DD*512);
    cvt_bf16_kernel<<<(DD*DD + 255)/256, 256, 0, stream>>>(wo, wob, DD*DD);

    // 3. Q/K/V projections (bf16 WMMA, bf16 outputs)
    gemm_bf16_kernel<<<dim3(NROWS/64, 2048/128), 256, 0, stream>>>(xn, wqb, qraw, NROWS, 2048, DD, 1);
    gemm_bf16_kernel<<<dim3(NROWS/64,  512/128), 256, 0, stream>>>(xn, wkb, kraw, NROWS,  512, DD, 1);
    gemm_bf16_kernel<<<dim3(NROWS/64,  512/128), 256, 0, stream>>>(xn, wvb, vraw, NROWS,  512, DD, 1);

    // 4. RoPE + head-major relayout
    rope_cvt_kernel<<<NROWS, 256, 0, stream>>>(qraw, qhm, HQn, 1);
    rope_cvt_kernel<<<NROWS, 256, 0, stream>>>(kraw, khm, HKVn, 1);
    rope_cvt_kernel<<<NROWS, 256, 0, stream>>>(vraw, vhm, HKVn, 0);

    // 5. flash sliding-window attention
    swa_kernel<<<BB*HQn*(SS/128), 256, 0, stream>>>(qhm, khm, vhm, attn);

    // 6. output projection -> f32 d_out
    gemm_bf16_kernel<<<dim3(NROWS/64, DD/128), 256, 0, stream>>>(attn, wob, d_out, NROWS, DD, DD, 0);
}
